// ArcFaceLoss_35424890257525
// MI455X (gfx1250) — matrix operand order repaired and checked by reading, not compiled, not run on hardware
//
#include <hip/hip_runtime.h>
#include <hip/hip_bf16.h>
#include <hip/hip_fp16.h>

// ---------------------------------------------------------------------------
// ArcFace loss, fully fused for MI455X (gfx1250, wave32, WMMA + TDM).
//   B=512 rows, F=128 inner dim, C=100000 classes.
//   loss = -mean_b( margin_b - log( sum_c exp(logit[b,c]) - exp(logit[b,t_b]) + exp(margin_b) ) )
// Main kernel: features f16 -> persistent A-fragments in VGPRs; w tiles DMA'd
// to LDS by the Tensor Data Mover (double buffered), converted to f16, fed to
// v_wmma_f32_16x16x32_f16; exp row-sums accumulate in registers.
// ---------------------------------------------------------------------------

typedef __attribute__((ext_vector_type(16))) _Float16     v16h;
typedef __attribute__((ext_vector_type(8)))  float        v8f;
typedef __attribute__((ext_vector_type(4)))  float        v4f;
typedef __attribute__((ext_vector_type(4)))  unsigned int v4u;
typedef __attribute__((ext_vector_type(8)))  unsigned int v8u;

#define B_ROWS   512
#define F_DIM    128
#define C_COLS   100000
#define NB       64                                    // columns per chunk
#define NCHUNK   ((C_COLS + NB - 1) / NB)              // 1563
#define GMAIN    192                                   // workgroups in main kernel
#define SA_STRIDE 136                                  // halfs per feature row (128 + pad)
#define SB_STRIDE 136                                  // halfs per w column (128 + pad)
#define SA_BYTES  (B_ROWS * SA_STRIDE * 2)             // 139264 B (transient)
// Reused region after A-frags become register resident:
#define RAW_BYTES (F_DIM * NB * 4)                     // 32768 B raw f32 tile
#define RAW_OFF0  0
#define RAW_OFF1  RAW_BYTES
#define SB_OFF0   (2 * RAW_BYTES)                      // 65536
#define SB_OFF1   (SB_OFF0 + NB * SB_STRIDE * 2)       // 65536 + 17408
#define SMEM_BYTES SA_BYTES                            // 139264 B dynamic LDS

union Frag {
    v16h h;
    v4u  q[2];
};

__device__ __forceinline__ void wait_tensorcnt0()
{
#if __has_builtin(__builtin_amdgcn_s_wait_tensorcnt)
    __builtin_amdgcn_s_wait_tensorcnt(0);
#else
    asm volatile("s_wait_tensorcnt 0x0" ::: "memory");
#endif
}

// Issue a TDM 2D tile load: global [128 rows x 64 cols f32, row stride C] ->
// contiguous LDS tile at byte offset lds_off. D# per CDNA5 ISA §8.3/8.4.
__device__ __forceinline__ void tdm_load_tile(const float* gptr, unsigned lds_off,
                                              unsigned dim0_rem)
{
    unsigned long long ga = (unsigned long long)(uintptr_t)gptr;
    v4u g0;
    g0[0] = 1u;                                           // count=1 (valid), user mode
    g0[1] = lds_off;                                      // lds_addr (bytes)
    g0[2] = (unsigned)ga;                                 // global_addr[31:0]
    g0[3] = (unsigned)((ga >> 32) & 0x01FFFFFFu)          // global_addr[56:32]
          | 0x80000000u;                                  // type = 2 ("image")
    v8u g1;
    g1[0] = (2u << 16);                                   // workgroup_mask=0, data_size=2 (4B)
    g1[1] = (dim0_rem & 0xFFFFu) << 16;                   // tensor_dim0[15:0]  @bits 63:48
    g1[2] = ((dim0_rem >> 16) & 0xFFFFu)                  // tensor_dim0[31:16] @bits 79:64
          | ((unsigned)F_DIM << 16);                      // tensor_dim1[15:0]=128 @bits 95:80
    g1[3] = ((unsigned)NB << 16);                         // tensor_dim1[31:16]=0, tile_dim0=64
    g1[4] = (unsigned)F_DIM;                              // tile_dim1=128, tile_dim2=0
    g1[5] = (unsigned)C_COLS;                             // tensor_dim0_stride[31:0]
    g1[6] = 0u;                                           // stride hi, tensor_dim1_stride lo
    g1[7] = 0u;
    asm volatile("tensor_load_to_lds %0, %1" :: "s"(g0), "s"(g1) : "memory");
}

// ---------------------------------------------------------------------------
// Kernel 1: per-sample prep (norms, target logit, ArcFace margin).
// ---------------------------------------------------------------------------
__global__ void __launch_bounds__(256)
arcface_prep(const float* __restrict__ feat, const float* __restrict__ w,
             const int* __restrict__ target,
             float* __restrict__ margin_out, float* __restrict__ explogit_out)
{
    int b = blockIdx.x * blockDim.x + threadIdx.x;
    if (b >= B_ROWS) return;

    const float* fr = feat + (size_t)b * F_DIM;
    int t = target[b];

    float fn2 = 0.f, wn2 = 0.f, dot = 0.f;
    #pragma unroll 8
    for (int k = 0; k < F_DIM; ++k) {
        float fv = fr[k];
        float wv = w[(size_t)k * C_COLS + t];
        fn2 += fv * fv;
        wn2 += wv * wv;
        dot += fv * wv;
    }
    float mod = sqrtf(fn2) * sqrtf(wn2);
    float ct  = dot / (mod * 1.01f);
    ct = fminf(1.f, fmaxf(-1.f, ct));
    float th  = acosf(ct) + 0.5f;               // ANGLE = 0.5
    margin_out[b]   = mod * cosf(th);
    explogit_out[b] = expf(dot);
}

// ---------------------------------------------------------------------------
// Kernel 2: fused GEMM + exp + row-sum with TDM staging.
// ---------------------------------------------------------------------------
__global__ void __launch_bounds__(256)
arcface_main(const float* __restrict__ feat, const float* __restrict__ w,
             float* __restrict__ partials)
{
    extern __shared__ char smem[];

    const int tid  = threadIdx.x;
    const int lane = tid & 31;
    const int wave = tid >> 5;        // 0..7
    const int l16  = lane & 15;
    const int kg   = lane >> 4;       // 0/1: 16-lane half-group

    // ---- stage features f32 -> f16 into sA (transient use of LDS) ----
    {
        _Float16* sA = (_Float16*)smem;
        #pragma unroll 4
        for (int it = 0; it < (B_ROWS * F_DIM / 4) / 256; ++it) {   // 64 iters
            int q   = it * 256 + tid;
            int row = q >> 5;
            int c4  = (q & 31) * 4;
            v4f v = *(const v4f*)(feat + (size_t)row * F_DIM + c4);
            _Float16* dst = sA + (size_t)row * SA_STRIDE + c4;
            dst[0] = (_Float16)v.x; dst[1] = (_Float16)v.y;
            dst[2] = (_Float16)v.z; dst[3] = (_Float16)v.w;
        }
    }
    __syncthreads();

    // ---- load this wave's 16 A-fragments into registers (persistent) ----
    Frag af[4][4];
    {
        const _Float16* sA = (const _Float16*)smem;
        #pragma unroll
        for (int m = 0; m < 4; ++m) {
            const int row = (wave * 4 + m) * 16 + l16;
            #pragma unroll
            for (int kk = 0; kk < 4; ++kk) {
                const _Float16* p = sA + (size_t)row * SA_STRIDE + kk * 32 + kg * 8;
                af[m][kk].q[0] = *(const v4u*)(p);       // K = kk*32 + kg*8 + 0..7
                af[m][kk].q[1] = *(const v4u*)(p + 16);  // K = kk*32 + 16 + kg*8 + 0..7
            }
        }
    }
    // All ds reads must land before the TDM overwrites the sA region.
    asm volatile("s_wait_dscnt 0x0" ::: "memory");
    __syncthreads();

    // ---- prologue: DMA first w tile ----
    if (wave == 0) {
        int c0 = blockIdx.x * NB;
        tdm_load_tile(w + (size_t)c0, RAW_OFF0, (unsigned)(C_COLS - c0));
    }

    float es[32];
    #pragma unroll
    for (int i = 0; i < 32; ++i) es[i] = 0.f;

    int ib = 0;
    for (int chunk = blockIdx.x; chunk < NCHUNK; chunk += GMAIN) {
        const int c0 = chunk * NB;
        const unsigned raw_off = ib ? RAW_OFF1 : RAW_OFF0;
        const unsigned sb_off  = ib ? SB_OFF1  : SB_OFF0;

        if (wave == 0) wait_tensorcnt0();
        __syncthreads();                    // raw tile visible to all waves

        // ---- convert raw f32 [128][64] -> sB f16 [64][SB_STRIDE] (transpose) ----
        {
            const float* rawp = (const float*)(smem + raw_off);
            _Float16*    sBp  = (_Float16*)(smem + sb_off);
            #pragma unroll
            for (int it = 0; it < (F_DIM * NB / 4) / 256; ++it) {   // 8 iters
                int q  = it * 256 + tid;
                int k  = q >> 4;
                int c4 = (q & 15) * 4;
                v4f v = *(const v4f*)(rawp + (size_t)k * NB + c4);
                _Float16* dst = sBp + (size_t)c4 * SB_STRIDE + k;
                dst[0 * SB_STRIDE] = (_Float16)v.x;
                dst[1 * SB_STRIDE] = (_Float16)v.y;
                dst[2 * SB_STRIDE] = (_Float16)v.z;
                dst[3 * SB_STRIDE] = (_Float16)v.w;
            }
        }
        __syncthreads();                    // sB ready; raw[ib] consumed

        // ---- prefetch next chunk via TDM (overlaps compute below) ----
        {
            int nxt = chunk + GMAIN;
            if (wave == 0 && nxt < NCHUNK) {
                int nc0 = nxt * NB;
                tdm_load_tile(w + (size_t)nc0, ib ? RAW_OFF0 : RAW_OFF1,
                              (unsigned)(C_COLS - nc0));
            }
        }

        // ---- compute: 4 n-tiles x 4 m-tiles, K = 128 = 4 x 32 ----
        {
            const _Float16* sBp = (const _Float16*)(smem + sb_off);
            #pragma unroll
            for (int nt = 0; nt < 4; ++nt) {
                Frag bf[4];
                #pragma unroll
                for (int kk = 0; kk < 4; ++kk) {
                    const _Float16* p = sBp + (size_t)(nt * 16 + l16) * SB_STRIDE
                                            + kk * 32 + kg * 8;
                    bf[kk].q[0] = *(const v4u*)(p);
                    bf[kk].q[1] = *(const v4u*)(p + 16);
                }
                const int  col = c0 + nt * 16 + l16;
                const bool ok  = (col < C_COLS);
                #pragma unroll
                for (int m = 0; m < 4; ++m) {
                    v8f acc = {0.f, 0.f, 0.f, 0.f, 0.f, 0.f, 0.f, 0.f};
                    #pragma unroll
                    for (int kk = 0; kk < 4; ++kk) {
                        acc = __builtin_amdgcn_wmma_f32_16x16x32_f16(
                                  false, af[m][kk].h, false, bf[kk].h,
                                  (short)0, acc, false, false);
                    }
                    #pragma unroll
                    for (int i = 0; i < 8; ++i) {
                        float e = ok ? __expf(acc[i]) : 0.f;
                        es[m * 8 + i] += e;
                    }
                }
            }
        }
        __syncthreads();                    // done with sB[ib] before re-fill
        ib ^= 1;
    }

    // ---- reduce exp-sums across the 16 column lanes, write per-block partials ----
    #pragma unroll
    for (int m = 0; m < 4; ++m) {
        #pragma unroll
        for (int i = 0; i < 8; ++i) {
            float v = es[m * 8 + i];
            v += __shfl_xor(v, 1, 32);
            v += __shfl_xor(v, 2, 32);
            v += __shfl_xor(v, 4, 32);
            v += __shfl_xor(v, 8, 32);
            if (l16 == 0) {
                int row = wave * 64 + m * 16 + kg * 8 + i;
                partials[(size_t)blockIdx.x * B_ROWS + row] = v;
            }
        }
    }
}

// ---------------------------------------------------------------------------
// Kernel 3: deterministic reduction of partials + final loss.
// ---------------------------------------------------------------------------
__global__ void __launch_bounds__(512)
arcface_finalize(const float* __restrict__ wsf, float* __restrict__ out)
{
    __shared__ float red[B_ROWS];
    const float* margin   = wsf;
    const float* explogit = wsf + B_ROWS;
    const float* partials = wsf + 2 * B_ROWS;

    int b = threadIdx.x;
    float S = 0.f;
    for (int g = 0; g < GMAIN; ++g) S += partials[(size_t)g * B_ROWS + b];

    float m    = margin[b];
    float top  = __expf(m);
    float down = S - explogit[b] + top;
    red[b] = m - __logf(down);     // log(top/down)
    __syncthreads();

    #pragma unroll
    for (int s = 256; s > 0; s >>= 1) {
        if (b < s) red[b] += red[b + s];
        __syncthreads();
    }
    if (b == 0) out[0] = -red[0] / (float)B_ROWS;
}

// ---------------------------------------------------------------------------
extern "C" void kernel_launch(void* const* d_in, const int* in_sizes, int n_in,
                              void* d_out, int out_size, void* d_ws, size_t ws_size,
                              hipStream_t stream)
{
    const float* feat   = (const float*)d_in[0];   // [512,128] f32
    const float* w      = (const float*)d_in[1];   // [128,100000] f32
    const int*   target = (const int*)d_in[2];     // [512]
    float* out = (float*)d_out;

    float* wsf      = (float*)d_ws;
    float* margin   = wsf;                 // 512 floats
    float* explogit = wsf + B_ROWS;        // 512 floats
    float* partials = wsf + 2 * B_ROWS;    // GMAIN*512 floats

    arcface_prep<<<2, 256, 0, stream>>>(feat, w, target, margin, explogit);
    arcface_main<<<GMAIN, 256, SMEM_BYTES, stream>>>(feat, w, partials);
    arcface_finalize<<<1, 512, 0, stream>>>(wsf, out);
}
